// SandboxLocalEventOperator_33346126086687
// MI455X (gfx1250) — matrix operand order repaired and checked by reading, not compiled, not run on hardware
//
#include <hip/hip_runtime.h>

#define NN 50000
#define NE 500000
#define H  128

typedef __attribute__((ext_vector_type(16))) _Float16 v16h;
typedef __attribute__((ext_vector_type(8)))  float    v8f;

__device__ __forceinline__ void lds_fence() {
  asm volatile("s_wait_dscnt 0" ::: "memory");
}

__device__ __forceinline__ v8f wmma16(v16h a, v16h b, v8f c) {
  // v_wmma_f32_16x16x32_f16  (neg_a, A, neg_b, B, c_mod, C, reuse_a, reuse_b)
  return __builtin_amdgcn_wmma_f32_16x16x32_f16(false, a, false, b, (short)0, c,
                                                false, false);
}

// A-fragment (16x32 f16) from an LDS row-major tile.
// ISA layout: lane<16 row M=lane holds K={0..7,16..23}; lane>=16 row M=lane-16
// holds K={8..15,24..31}; VGPR v packs K pairs.
__device__ __forceinline__ v16h load_afrag(const _Float16* tile, int stride,
                                           int kbase, int lane) {
  const _Float16* row = tile + (size_t)(lane & 15) * stride + kbase +
                        ((lane >= 16) ? 8 : 0);
  v16h a;
#pragma unroll
  for (int h = 0; h < 16; ++h) {
    int p = h >> 1, o = h & 1;
    int k = ((p >> 2) << 4) + ((p & 3) << 1) + o;
    a[h] = row[k];
  }
  return a;
}

// One 16-row x (16*NTILES)-col output tile: D = A(16xK) * W(KxNout) + bias.
// B fragments come pre-packed: frag (ks,nt) = 32 lanes x 16 halfs contiguous.
template <int KSTEPS, int NTILES>
__device__ __forceinline__ void gemm_tile(const _Float16* atile, int stride,
                                          const _Float16* wp, const float* bias,
                                          int nw, int lane, v8f d[NTILES]) {
  int nc = lane & 15;
#pragma unroll
  for (int nt = 0; nt < NTILES; ++nt) {
    int n = nt * 16 + nc;
    float bv = (n < nw) ? bias[n] : 0.0f;
    v8f c;
#pragma unroll
    for (int r = 0; r < 8; ++r) c[r] = bv;
    d[nt] = c;
  }
#pragma unroll
  for (int ks = 0; ks < KSTEPS; ++ks) {
    v16h a = load_afrag(atile, stride, ks * 32, lane);
#pragma unroll
    for (int nt = 0; nt < NTILES; ++nt) {
      v16h b = *(const v16h*)(wp + (size_t)(ks * NTILES + nt) * 512 + lane * 16);
      d[nt] = wmma16(a, b, d[nt]);
    }
  }
}

// Write a D fragment (relu'd, f16) back into an LDS row-major tile.
__device__ __forceinline__ void store_frag_relu(_Float16* tile, int stride,
                                                int nt, int lane, v8f d) {
  int n = nt * 16 + (lane & 15);
  int mb = (lane >= 16) ? 8 : 0;
#pragma unroll
  for (int r = 0; r < 8; ++r) {
    float v = d[r];
    v = v > 0.0f ? v : 0.0f;
    tile[(size_t)(mb + r) * stride + n] = (_Float16)v;
  }
}

// ---------------------------------------------------------------------------
// Weight packer: f32 row-major [K x Nw] -> f16 B-fragment stream.
// frag index = ks*ntiles + nt ; entry (lane,h): k = ks*32 + (lane>=16?8:0) + (h<8 ? h : h+8),
// n = nt*16 + (lane&15). perm=1 applies the edge-context column reorder used
// for We1 (src|dst|ef|emb|params|tgt|scope).
__global__ void pack_w(const float* __restrict__ W, _Float16* __restrict__ Wp,
                       int K, int Nw, int ksteps, int ntiles, int perm) {
  int total = ksteps * ntiles * 512;
  for (int i = blockIdx.x * blockDim.x + threadIdx.x; i < total;
       i += gridDim.x * blockDim.x) {
    int h = i & 15;
    int lane = (i >> 4) & 31;
    int nt = (i >> 9) % ntiles;
    int ks = (i >> 9) / ntiles;
    int kp = ks * 32 + ((lane >= 16) ? 8 : 0) + ((h < 8) ? h : h + 8);
    int n = nt * 16 + (lane & 15);
    int ko = kp;
    if (perm) ko = (kp < 256) ? kp + 4 : (kp < 260 ? kp - 256 : kp);
    float v = 0.0f;
    if (ko < K && n < Nw) v = W[(size_t)ko * Nw + n];
    Wp[i] = (_Float16)v;
  }
}

// ---------------------------------------------------------------------------
// Kernel 1: node MLP (ctx[29 -> pad 32] -> H -> H), writes node_h (f16) and
// zeroes the aggregation buffer.
__global__ __launch_bounds__(128) void k_node_mlp(
    const float* __restrict__ nf, const float* __restrict__ emb,
    const int* __restrict__ etype, const float* __restrict__ eparams,
    const int* __restrict__ nmask, const int* __restrict__ nscope,
    const int* __restrict__ nbatch, const _Float16* __restrict__ wn1p,
    const float* __restrict__ bn1, const _Float16* __restrict__ wn2p,
    const float* __restrict__ bn2, _Float16* __restrict__ node_h,
    float* __restrict__ agg) {
  __shared__ __align__(16) _Float16 sctx[4][16 * 32];
  __shared__ __align__(16) _Float16 sh1[4][16 * 128];
  int lane = threadIdx.x & 31, wv = threadIdx.x >> 5;
  int tile = blockIdx.x * 4 + wv;
  if (tile >= NN / 16) return;
  int base = tile * 16;
  _Float16* ctx = sctx[wv];
  _Float16* h1 = sh1[wv];
  {
    int m = lane & 15;
    int n = base + m;
    int b = nbatch[n];
    int et = etype[b];
    int c0 = (lane >= 16) ? 16 : 0;
    for (int c = c0; c < c0 + 16; ++c) {
      float v;
      if (c < 7) v = nf[(size_t)n * 7 + c];
      else if (c < 23) v = emb[et * 16 + (c - 7)];
      else if (c < 27) v = eparams[b * 4 + (c - 23)];
      else if (c == 27) v = (float)nmask[n];
      else if (c == 28) v = (float)nscope[n];
      else v = 0.0f;
      ctx[m * 32 + c] = (_Float16)v;
    }
  }
  lds_fence();
  v8f d1[8];
  gemm_tile<1, 8>(ctx, 32, wn1p, bn1, H, lane, d1);
#pragma unroll
  for (int nt = 0; nt < 8; ++nt) store_frag_relu(h1, 128, nt, lane, d1[nt]);
  lds_fence();
  v8f d2[8];
  gemm_tile<4, 8>(h1, 128, wn2p, bn2, H, lane, d2);
  lds_fence();
#pragma unroll
  for (int nt = 0; nt < 8; ++nt) store_frag_relu(h1, 128, nt, lane, d2[nt]);
  lds_fence();
  // coalesced store of the 16x128 f16 tile + zero the aggregation rows
  const uint4* s = (const uint4*)h1;
  uint4* g = (uint4*)(node_h + (size_t)base * H);
  for (int i = lane; i < 256; i += 32) g[i] = s[i];
  float4 z = make_float4(0.f, 0.f, 0.f, 0.f);
  float4* ag = (float4*)(agg + (size_t)base * H);
  for (int i = lane; i < 512; i += 32) ag[i] = z;
}

// ---------------------------------------------------------------------------
// Kernel 2: edge MLP (ctx 288 -> H -> H), messages (H->H), scatter-add.
__global__ __launch_bounds__(128) void k_edge(
    const float* __restrict__ ef, const int* __restrict__ eidx,
    const float* __restrict__ emb, const int* __restrict__ etype,
    const float* __restrict__ eparams, const int* __restrict__ emask,
    const int* __restrict__ escope, const int* __restrict__ ebatch,
    const _Float16* __restrict__ node_h, const _Float16* __restrict__ we1p,
    const float* __restrict__ be1, const _Float16* __restrict__ we2p,
    const float* __restrict__ be2, const _Float16* __restrict__ wmp,
    const float* __restrict__ bm, _Float16* __restrict__ edge_h,
    float* __restrict__ agg) {
  __shared__ __align__(16) _Float16 sctx[4][16 * 288];
  __shared__ int ssrc[4][16], sdst[4][16];
  __shared__ float sscope[4][16];
  int lane = threadIdx.x & 31, wv = threadIdx.x >> 5;
  int tile = blockIdx.x * 4 + wv;
  if (tile >= NE / 16) return;
  int base = tile * 16;
  _Float16* ctx = sctx[wv];
  {
    int m = lane & 15;
    int e = base + m;
    int s = eidx[e], d = eidx[NE + e];
    uint4* crow = (uint4*)(ctx + (size_t)m * 288);
    if (lane < 16) {
      const uint4* sr = (const uint4*)(node_h + (size_t)s * H);
#pragma unroll
      for (int i = 0; i < 16; ++i) crow[i] = sr[i];
      int b = ebatch[e], et = etype[b];
      _Float16* t = ctx + (size_t)m * 288 + 256;
      for (int c = 0; c < 4; ++c) t[c] = (_Float16)ef[(size_t)e * 4 + c];
      for (int c = 0; c < 16; ++c) t[4 + c] = (_Float16)emb[et * 16 + c];
      for (int c = 0; c < 4; ++c) t[20 + c] = (_Float16)eparams[b * 4 + c];
      t[24] = (_Float16)(float)emask[e];
      float sc = (float)escope[e];
      t[25] = (_Float16)sc;
      for (int c = 26; c < 32; ++c) t[c] = (_Float16)0.0f;
      ssrc[wv][m] = s;
      sdst[wv][m] = d;
      sscope[wv][m] = sc;
    } else {
      const uint4* dr = (const uint4*)(node_h + (size_t)d * H);
#pragma unroll
      for (int i = 0; i < 16; ++i) crow[16 + i] = dr[i];
    }
  }
  lds_fence();
  v8f dfr[8];
  gemm_tile<9, 8>(ctx, 288, we1p, be1, H, lane, dfr);
  lds_fence();
#pragma unroll
  for (int nt = 0; nt < 8; ++nt) store_frag_relu(ctx, 288, nt, lane, dfr[nt]);
  lds_fence();
  gemm_tile<4, 8>(ctx, 288, we2p, be2, H, lane, dfr);
  lds_fence();
#pragma unroll
  for (int nt = 0; nt < 8; ++nt) store_frag_relu(ctx, 288, nt, lane, dfr[nt]);
  lds_fence();
  {
    int m = lane & 15;
    uint4* gr = (uint4*)(edge_h + (size_t)(base + m) * H);
    const uint4* cr = (const uint4*)(ctx + (size_t)m * 288);
    int o = (lane < 16) ? 0 : 8;
#pragma unroll
    for (int i = 0; i < 8; ++i) gr[o + i] = cr[o + i];
  }
  v8f mfr[8];
  gemm_tile<4, 8>(ctx, 288, wmp, bm, H, lane, mfr);
  int nc = lane & 15, mb = (lane >= 16) ? 8 : 0;
#pragma unroll
  for (int nt = 0; nt < 8; ++nt) {
#pragma unroll
    for (int r = 0; r < 8; ++r) {
      int row = mb + r;
      float sc = sscope[wv][row];
      if (sc != 0.0f) {
        float v = mfr[nt][r] * sc;
        int s = ssrc[wv][row], d = sdst[wv][row];
        atomicAdd(agg + (size_t)s * H + nt * 16 + nc, v);
        atomicAdd(agg + (size_t)d * H + nt * 16 + nc, v);
      }
    }
  }
}

// ---------------------------------------------------------------------------
// Kernel 3: node update ([node_h|agg] 256 -> H), node delta (H -> 7), outputs.
__global__ __launch_bounds__(128) void k_node_upd(
    const float* __restrict__ nf, const int* __restrict__ nscope,
    const _Float16* __restrict__ node_h, const float* __restrict__ agg,
    const _Float16* __restrict__ wup, const float* __restrict__ bu,
    const _Float16* __restrict__ wndp, const float* __restrict__ bnd,
    _Float16* __restrict__ node_h_upd, float* __restrict__ out_nd,
    float* __restrict__ out_nfp) {
  __shared__ __align__(16) _Float16 sctx[4][16 * 256];
  int lane = threadIdx.x & 31, wv = threadIdx.x >> 5;
  int tile = blockIdx.x * 4 + wv;
  if (tile >= NN / 16) return;
  int base = tile * 16;
  _Float16* ctx = sctx[wv];
  int m = lane & 15;
  int n = base + m;
  if (lane < 16) {
    uint4* crow = (uint4*)(ctx + (size_t)m * 256);
    const uint4* hr = (const uint4*)(node_h + (size_t)n * H);
#pragma unroll
    for (int i = 0; i < 16; ++i) crow[i] = hr[i];
  } else {
    const float* ar = agg + (size_t)n * H;
    _Float16* c = ctx + (size_t)m * 256 + 128;
    for (int i = 0; i < 128; ++i) c[i] = (_Float16)ar[i];
  }
  lds_fence();
  v8f dfr[8];
  gemm_tile<8, 8>(ctx, 256, wup, bu, H, lane, dfr);
  lds_fence();
#pragma unroll
  for (int nt = 0; nt < 8; ++nt) store_frag_relu(ctx, 256, nt, lane, dfr[nt]);
  lds_fence();
  {
    uint4* gr = (uint4*)(node_h_upd + (size_t)n * H);
    const uint4* cr = (const uint4*)(ctx + (size_t)m * 256);
    int o = (lane < 16) ? 0 : 8;
#pragma unroll
    for (int i = 0; i < 8; ++i) gr[o + i] = cr[o + i];
  }
  v8f dd[1];
  gemm_tile<4, 1>(ctx, 256, wndp, bnd, 7, lane, dd);
  int nc = lane & 15, mb = (lane >= 16) ? 8 : 0;
  if (nc < 7) {
#pragma unroll
    for (int r = 0; r < 8; ++r) {
      int nn = base + mb + r;
      float sc = (float)nscope[nn];
      float raw = dd[0][r];
      out_nd[(size_t)nn * 7 + nc] = (sc != 0.0f) ? raw : 0.0f;
      float f = nf[(size_t)nn * 7 + nc];
      out_nfp[(size_t)nn * 7 + nc] = (sc != 0.0f) ? (f + raw) : f;
    }
  }
}

// ---------------------------------------------------------------------------
// Kernel 4: edge delta ([edge_h|upd_src|upd_dst] 384 -> H -> 4), outputs.
__global__ __launch_bounds__(128) void k_edge_delta(
    const float* __restrict__ ef, const int* __restrict__ eidx,
    const int* __restrict__ escope, const _Float16* __restrict__ edge_h,
    const _Float16* __restrict__ node_h_upd, const _Float16* __restrict__ wed1p,
    const float* __restrict__ bed1, const _Float16* __restrict__ wed2p,
    const float* __restrict__ bed2, float* __restrict__ out_ed,
    float* __restrict__ out_efp) {
  __shared__ __align__(16) _Float16 sctx[4][16 * 384];
  int lane = threadIdx.x & 31, wv = threadIdx.x >> 5;
  int tile = blockIdx.x * 4 + wv;
  if (tile >= NE / 16) return;
  int base = tile * 16;
  _Float16* ctx = sctx[wv];
  int m = lane & 15;
  int e = base + m;
  {
    int s = eidx[e], d = eidx[NE + e];
    uint4* crow = (uint4*)(ctx + (size_t)m * 384);
    const uint4* eh = (const uint4*)(edge_h + (size_t)e * H);
    const uint4* sr = (const uint4*)(node_h_upd + (size_t)s * H);
    const uint4* dr = (const uint4*)(node_h_upd + (size_t)d * H);
    if (lane < 16) {
#pragma unroll
      for (int i = 0; i < 16; ++i) crow[i] = eh[i];
#pragma unroll
      for (int i = 0; i < 8; ++i) crow[16 + i] = sr[i];
    } else {
#pragma unroll
      for (int i = 0; i < 8; ++i) crow[24 + i] = sr[8 + i];
#pragma unroll
      for (int i = 0; i < 16; ++i) crow[32 + i] = dr[i];
    }
  }
  lds_fence();
  v8f dfr[8];
  gemm_tile<12, 8>(ctx, 384, wed1p, bed1, H, lane, dfr);
  lds_fence();
#pragma unroll
  for (int nt = 0; nt < 8; ++nt) store_frag_relu(ctx, 384, nt, lane, dfr[nt]);
  lds_fence();
  v8f dd[1];
  gemm_tile<4, 1>(ctx, 384, wed2p, bed2, 4, lane, dd);
  int nc = lane & 15, mb = (lane >= 16) ? 8 : 0;
  if (nc < 4) {
#pragma unroll
    for (int r = 0; r < 8; ++r) {
      int ee = base + mb + r;
      float sc = (float)escope[ee];
      float raw = dd[0][r];
      out_ed[(size_t)ee * 4 + nc] = (sc != 0.0f) ? raw : 0.0f;
      float f = ef[(size_t)ee * 4 + nc];
      out_efp[(size_t)ee * 4 + nc] = (sc != 0.0f) ? (f + raw) : f;
    }
  }
}

// ---------------------------------------------------------------------------
extern "C" void kernel_launch(void* const* d_in, const int* in_sizes, int n_in,
                              void* d_out, int out_size, void* d_ws,
                              size_t ws_size, hipStream_t stream) {
  const float* nf = (const float*)d_in[0];
  const int* eidx = (const int*)d_in[1];
  const float* ef = (const float*)d_in[2];
  const int* etype = (const int*)d_in[3];
  const float* eparams = (const float*)d_in[4];
  const int* nmask = (const int*)d_in[5];
  const int* emask = (const int*)d_in[6];
  const int* nscope = (const int*)d_in[7];
  const int* escope = (const int*)d_in[8];
  const int* nbatch = (const int*)d_in[9];
  const int* ebatch = (const int*)d_in[10];
  const float* emb = (const float*)d_in[13];
  const float* Wn1 = (const float*)d_in[14];
  const float* bn1 = (const float*)d_in[15];
  const float* Wn2 = (const float*)d_in[16];
  const float* bn2 = (const float*)d_in[17];
  const float* We1 = (const float*)d_in[18];
  const float* be1 = (const float*)d_in[19];
  const float* We2 = (const float*)d_in[20];
  const float* be2 = (const float*)d_in[21];
  const float* Wm = (const float*)d_in[22];
  const float* bm = (const float*)d_in[23];
  const float* Wu = (const float*)d_in[24];
  const float* bu = (const float*)d_in[25];
  const float* Wnd = (const float*)d_in[26];
  const float* bnd = (const float*)d_in[27];
  const float* Wed1 = (const float*)d_in[28];
  const float* bed1 = (const float*)d_in[29];
  const float* Wed2 = (const float*)d_in[30];
  const float* bed2 = (const float*)d_in[31];

  float* out = (float*)d_out;
  float* out_nd = out;
  float* out_ed = out + (size_t)NN * 7;
  float* out_nfp = out_ed + (size_t)NE * 4;
  float* out_efp = out_nfp + (size_t)NN * 7;

  char* ws = (char*)d_ws;
  size_t off = 0;
  auto alloc = [&](size_t bytes) -> void* {
    off = (off + 255) & ~(size_t)255;
    void* p = ws + off;
    off += bytes;
    return p;
  };
  float* agg = (float*)alloc((size_t)NN * H * 4);
  _Float16* node_h = (_Float16*)alloc((size_t)NN * H * 2);
  _Float16* node_h_upd = (_Float16*)alloc((size_t)NN * H * 2);
  _Float16* edge_h = (_Float16*)alloc((size_t)NE * H * 2);
  _Float16* wn1p = (_Float16*)alloc((size_t)1 * 8 * 512 * 2);
  _Float16* wn2p = (_Float16*)alloc((size_t)4 * 8 * 512 * 2);
  _Float16* we1p = (_Float16*)alloc((size_t)9 * 8 * 512 * 2);
  _Float16* we2p = (_Float16*)alloc((size_t)4 * 8 * 512 * 2);
  _Float16* wmp = (_Float16*)alloc((size_t)4 * 8 * 512 * 2);
  _Float16* wup = (_Float16*)alloc((size_t)8 * 8 * 512 * 2);
  _Float16* wndp = (_Float16*)alloc((size_t)4 * 1 * 512 * 2);
  _Float16* wed1p = (_Float16*)alloc((size_t)12 * 8 * 512 * 2);
  _Float16* wed2p = (_Float16*)alloc((size_t)4 * 1 * 512 * 2);

  auto pk = [&](const float* W, _Float16* Wp, int K, int Nw, int ks, int nt,
                int perm) {
    int total = ks * nt * 512;
    pack_w<<<(total + 255) / 256, 256, 0, stream>>>(W, Wp, K, Nw, ks, nt, perm);
  };
  pk(Wn1, wn1p, 29, 128, 1, 8, 0);
  pk(Wn2, wn2p, 128, 128, 4, 8, 0);
  pk(We1, we1p, 282, 128, 9, 8, 1);
  pk(We2, we2p, 128, 128, 4, 8, 0);
  pk(Wm, wmp, 128, 128, 4, 8, 0);
  pk(Wu, wup, 256, 128, 8, 8, 0);
  pk(Wnd, wndp, 128, 7, 4, 1, 0);
  pk(Wed1, wed1p, 384, 128, 12, 8, 0);
  pk(Wed2, wed2p, 128, 4, 4, 1, 0);

  int node_tiles = NN / 16;  // 3125
  int edge_tiles = NE / 16;  // 31250
  dim3 blk(128);
  k_node_mlp<<<(node_tiles + 3) / 4, blk, 0, stream>>>(
      nf, emb, etype, eparams, nmask, nscope, nbatch, wn1p, bn1, wn2p, bn2,
      node_h, agg);
  k_edge<<<(edge_tiles + 3) / 4, blk, 0, stream>>>(
      ef, eidx, emb, etype, eparams, emask, escope, ebatch, node_h, we1p, be1,
      we2p, be2, wmp, bm, edge_h, agg);
  k_node_upd<<<(node_tiles + 3) / 4, blk, 0, stream>>>(
      nf, nscope, node_h, agg, wup, bu, wndp, bnd, node_h_upd, out_nd, out_nfp);
  k_edge_delta<<<(edge_tiles + 3) / 4, blk, 0, stream>>>(
      ef, eidx, escope, edge_h, node_h_upd, wed1p, bed1, wed2p, bed2, out_ed,
      out_efp);
}
